// MemoryAugmentedGraphModel_37967510896686
// MI455X (gfx1250) — matrix operand order repaired and checked by reading, not compiled, not run on hardware
//
#include <hip/hip_runtime.h>
#include <math.h>

// ---------------------------------------------------------------------------
// Types for CDNA5 WMMA (gfx1250, wave32)
// ---------------------------------------------------------------------------
typedef __attribute__((ext_vector_type(16))) __bf16 v16bf;
typedef __attribute__((ext_vector_type(8)))  float  v8f;

static __device__ inline v8f vzero8() {
  v8f z;
#pragma unroll
  for (int i = 0; i < 8; ++i) z[i] = 0.0f;
  return z;
}

static __device__ inline float sigf(float x) { return 1.0f / (1.0f + __expf(-x)); }

#define WMMA_BF16(a, b, c) \
  __builtin_amdgcn_wmma_f32_16x16x32_bf16(false, (a), false, (b), (short)0, (c), false, false)

// ---------------------------------------------------------------------------
// Model dims
// ---------------------------------------------------------------------------
#define BB    256
#define SS    64
#define NL    10000
#define DD    256
#define SLOTS 512
#define RANKP 64      // RANK=50 padded to 64
#define NKPAD 10048   // NL padded to 157*64
#define NCHUNK 157

// Swizzled bf16 layout so a WMMA B-fragment is one contiguous 32B run/lane:
// elem(j,n) -> (((j>>5)*16 + (n>>4))*2 + ((j>>4)&1))*256 + (n&15)*16 + (j&15)
static __device__ inline size_t swz_off(int j, int n) {
  return ((((size_t)(j >> 5)) * 16 + (n >> 4)) * 2 + ((j >> 4) & 1)) * 256 +
         (size_t)(n & 15) * 16 + (j & 15);
}

// ===========================================================================
// Generic WMMA GEMM:  C[m][n] = scale * sum_k A[m][k] * B(k,n)  (+bias+add, act)
//   btrans=1 : B is W-layout (N x K row-major)  -> B(k,n) = B[n*ldb + k]
//   btrans=0 : B is K x N row-major             -> B(k,n) = B[k*ldb + n]
// Wave computes one 16 x 64 tile.  FULL=true: all 4 n-tiles valid (N%64 body).
// M % 16 == 0, K % 32 == 0 required.  act: 0=none 1=relu 2=gelu(exact)
// swz: write outH in swizzled B-fragment layout instead of row-major.
// ===========================================================================
template <bool FULL>
__global__ __launch_bounds__(32) void gemm_wmma(
    const float* __restrict__ A, int lda,
    const float* __restrict__ B, int ldb, int btrans,
    const float* __restrict__ bias,
    const float* __restrict__ add, int add_ld,
    float scale, int act,
    float* __restrict__ outF, __bf16* __restrict__ outH, int swz, int ldo,
    int n0base, int N, int K) {
  const int m0 = blockIdx.x * 16;
  const int n0 = n0base + blockIdx.y * 64;
  const int lane = threadIdx.x;
  const int lr = lane & 15;
  const int hi = lane >> 4;

  v8f acc[4];
#pragma unroll
  for (int t = 0; t < 4; ++t) acc[t] = vzero8();

  for (int k0 = 0; k0 < K; k0 += 32) {
    // A fragment (16x32 bf16): lanes<16: K k0+0..15 (e0..15)
    //                          lanes>=16: K k0+16..31
    v16bf a;
    {
      const float* p = A + (size_t)(m0 + lr) * lda + k0 + hi * 16;
#pragma unroll
      for (int i = 0; i < 8; ++i) {
        a[i]     = (__bf16)p[i];
        a[i + 8] = (__bf16)p[i + 8];
      }
      if (k0 + 32 < K) __builtin_prefetch(p + 32, 0, 3);
    }
#pragma unroll
    for (int t = 0; t < 4; ++t) {
      const int nc = n0 + 16 * t;
      if (FULL || nc < N) {
        // B fragment (32x16): lane col = lr, K = k0 + hi*16 + 0..15
        v16bf b;
        if (btrans) {
          const float* p = B + (size_t)(nc + lr) * ldb + k0 + hi * 16;
#pragma unroll
          for (int i = 0; i < 16; ++i) b[i] = (__bf16)p[i];
        } else {
          const float* p = B + (size_t)(k0 + hi * 16) * ldb + nc + lr;
#pragma unroll
          for (int i = 0; i < 16; ++i) b[i] = (__bf16)p[(size_t)i * ldb];
        }
        acc[t] = WMMA_BF16(a, b, acc[t]);
      }
    }
  }

#pragma unroll
  for (int t = 0; t < 4; ++t) {
    const int nc = n0 + 16 * t;
    if (FULL || nc < N) {
      const int col = nc + lr;
#pragma unroll
      for (int r = 0; r < 8; ++r) {
        const int row = m0 + r + 8 * hi;
        float v = acc[t][r] * scale;
        if (bias) v += bias[col];
        if (add)  v += add[(size_t)row * add_ld + col];
        if (act == 1) v = fmaxf(v, 0.0f);
        else if (act == 2) v = 0.5f * v * (1.0f + erff(v * 0.70710678118654752f));
        if (outF) outF[(size_t)row * ldo + col] = v;
        if (outH) {
          if (swz) outH[swz_off(row, col)] = (__bf16)v;
          else     outH[(size_t)row * ldo + col] = (__bf16)v;
        }
      }
    }
  }
}

// ===========================================================================
// Flash softmax-matmul for the GCN adjacency:
//   out = softmax(Lp @ Rt^T) @ V  (+bias, opt relu)
//   Lp : NL x 64 bf16 (adj_left K-padded)    Rt : NKPAD x 64 bf16 (adj_right^T)
//   V  : swizzled bf16 (NKPAD x 256)         out: NL x 256 f32
// One wave handles 16 rows x 256 cols, streaming 157 chunks of 64 keys with
// online softmax.  Rt chunks are double-buffered into LDS with
// global_load_async_to_lds_b128 (16 instrs/chunk, ASYNCcnt<=32).
// ===========================================================================
__global__ __launch_bounds__(32) void adj_flash(
    const __bf16* __restrict__ Lp, const __bf16* __restrict__ Rt,
    const __bf16* __restrict__ V, const float* __restrict__ bias,
    float* __restrict__ out, int relu) {
  __shared__ __bf16 rtbuf[2][64 * RANKP];   // 2 x 8KB
  __shared__ __bf16 pbuf[16 * 64];          // 2KB

  const int lane = threadIdx.x;
  const int lr = lane & 15;
  const int hi = lane >> 4;
  const int m0 = blockIdx.x * 16;

  // LDS byte offsets of the two Rt buffers (generic ptr low 32 bits = LDS offset)
  const unsigned rtlds0 = (unsigned)(size_t)(&rtbuf[0][0]);
  const unsigned rtlds1 = (unsigned)(size_t)(&rtbuf[1][0]);

  // L A-fragments (K = 64 -> 2 fragments), constant over all chunks
  v16bf aL[2];
#pragma unroll
  for (int ks = 0; ks < 2; ++ks) {
    const __bf16* p = Lp + (size_t)(m0 + lr) * RANKP + ks * 32 + hi * 16;
#pragma unroll
    for (int i = 0; i < 8; ++i) {
      aL[ks][i]     = p[i];
      aL[ks][i + 8] = p[i + 8];
    }
  }

  v8f acc[16];
#pragma unroll
  for (int t = 0; t < 16; ++t) acc[t] = vzero8();
  float rmax[8], rsum[8];
#pragma unroll
  for (int r = 0; r < 8; ++r) { rmax[r] = -3.0e38f; rsum[r] = 0.0f; }

  // issue async staging of one 8KB Rt chunk (16 b128 instructions)
  auto issue_chunk = [&](int ci, unsigned ldsbase) {
    const char* src = (const char*)Rt + (size_t)ci * 64 * RANKP * 2;
#pragma unroll
    for (int k = 0; k < 16; ++k) {
      unsigned loff = ldsbase + (unsigned)((k * 32 + lane) * 16);
      const char* g = src + (size_t)(k * 32 + lane) * 16;
      asm volatile("global_load_async_to_lds_b128 %0, %1, off"
                   :: "v"(loff), "v"(g) : "memory");
    }
  };

  issue_chunk(0, rtlds0);

  for (int ci = 0; ci < NCHUNK; ++ci) {
    const int j0 = ci * 64;
    const __bf16* rcur = &rtbuf[ci & 1][0];

    if (ci + 1 < NCHUNK) {
      issue_chunk(ci + 1, ((ci + 1) & 1) ? rtlds1 : rtlds0);
      asm volatile("s_wait_asynccnt 0x10" ::: "memory");  // chunk ci complete
    } else {
      asm volatile("s_wait_asynccnt 0x0" ::: "memory");
    }

    // ---- logits tile S = Lp @ Rt^T  (16 x 64), keys from LDS ----
    v8f st[4];
#pragma unroll
    for (int t = 0; t < 4; ++t) st[t] = vzero8();
#pragma unroll
    for (int t = 0; t < 4; ++t) {
#pragma unroll
      for (int ks = 0; ks < 2; ++ks) {
        const v16bf* bp =
            (const v16bf*)(rcur + (size_t)(16 * t + lr) * RANKP + ks * 32 + hi * 16);
        st[t] = WMMA_BF16(aL[ks], *bp, st[t]);
      }
    }

    // ---- online softmax (lanes0-15 -> row r, lanes16-31 -> row r+8) ----
#pragma unroll
    for (int r = 0; r < 8; ++r) {
      float mx = rmax[r];
#pragma unroll
      for (int t = 0; t < 4; ++t) {
        const int j = j0 + 16 * t + lr;
        float s = (j < NL) ? st[t][r] : -3.0e38f;
        st[t][r] = s;
        mx = fmaxf(mx, s);
      }
#pragma unroll
      for (int d = 1; d < 16; d <<= 1) mx = fmaxf(mx, __shfl_xor(mx, d, 32));
      const float sc = __expf(rmax[r] - mx);
      rmax[r] = mx;
      float ls = 0.0f;
#pragma unroll
      for (int t = 0; t < 4; ++t) {
        const float pv = __expf(st[t][r] - mx);
        st[t][r] = pv;
        ls += pv;
      }
#pragma unroll
      for (int d = 1; d < 16; d <<= 1) ls += __shfl_xor(ls, d, 32);
      rsum[r] = rsum[r] * sc + ls;
#pragma unroll
      for (int t = 0; t < 16; ++t) acc[t][r] *= sc;
    }

    // ---- stage P (16x64) to LDS, reload as A-fragments ----
    __syncthreads();
#pragma unroll
    for (int r = 0; r < 8; ++r)
#pragma unroll
      for (int t = 0; t < 4; ++t)
        pbuf[(r + 8 * hi) * 64 + 16 * t + lr] = (__bf16)st[t][r];
    __syncthreads();

    v16bf ap[2];
#pragma unroll
    for (int ks = 0; ks < 2; ++ks) {
      const __bf16* p = &pbuf[lr * 64 + ks * 32 + hi * 16];
#pragma unroll
      for (int i = 0; i < 8; ++i) {
        ap[ks][i]     = p[i];
        ap[ks][i + 8] = p[i + 8];
      }
    }

    // ---- acc += P @ V_chunk : V fragments are contiguous 32B in swizzled V --
    const size_t blk = (size_t)ci * 2;
#pragma unroll
    for (int t16 = 0; t16 < 16; ++t16) {
#pragma unroll
      for (int ks = 0; ks < 2; ++ks) {
        const v16bf* vp = (const v16bf*)(V + (((blk + ks) * 16 + t16) * 2 + hi) * 256 +
                                         (size_t)lr * 16);
        acc[t16] = WMMA_BF16(ap[ks], *vp, acc[t16]);
      }
    }
  }

  // ---- epilogue ----
#pragma unroll
  for (int t = 0; t < 16; ++t) {
    const int col = 16 * t + lr;
#pragma unroll
    for (int r = 0; r < 8; ++r) {
      const int row = m0 + r + 8 * hi;
      float v = acc[t][r] / rsum[r] + bias[col];
      if (relu) v = fmaxf(v, 0.0f);
      out[(size_t)row * DD + col] = v;
    }
  }
}

// ===========================================================================
// Small support kernels
// ===========================================================================
__global__ void pad_left_k(const float* __restrict__ L, __bf16* __restrict__ Lp) {
  int idx = blockIdx.x * 256 + threadIdx.x;
  if (idx >= NL * RANKP) return;
  int i = idx / RANKP, k = idx % RANKP;
  Lp[idx] = (k < 50) ? (__bf16)L[i * 50 + k] : (__bf16)0.0f;
}

__global__ void pad_right_k(const float* __restrict__ R, __bf16* __restrict__ Rt) {
  int idx = blockIdx.x * 256 + threadIdx.x;
  if (idx >= NKPAD * RANKP) return;
  int j = idx / RANKP, k = idx % RANKP;
  Rt[idx] = (j < NL && k < 50) ? (__bf16)R[(size_t)k * NL + j] : (__bf16)0.0f;
}

__global__ void addbias_k(const float* __restrict__ a, const float* __restrict__ b,
                          float* __restrict__ o, int n) {
  int i = blockIdx.x * 256 + threadIdx.x;
  if (i < n) o[i] = a[i] + b[i];
}

__global__ __launch_bounds__(256) void embed_k(
    const int* __restrict__ loc, const int* __restrict__ users,
    const int* __restrict__ smin, const int* __restrict__ wd,
    const float* __restrict__ be, const float* __restrict__ nb2,
    const float* __restrict__ ue, const float* __restrict__ he,
    const float* __restrict__ we, float* __restrict__ xseq) {
  int bs = blockIdx.x, d = threadIdx.x;
  int b = bs >> 6;
  int l = loc[bs];
  int u = users[b * SS];
  int hr = (smin[bs] / 60) % 24;
  int w = wd[bs];
  xseq[(size_t)bs * DD + d] = be[(size_t)l * DD + d] + nb2[(size_t)l * DD + d] +
                              ue[(size_t)u * DD + d] + he[hr * DD + d] + we[w * DD + d];
}

__global__ __launch_bounds__(256) void lstm_cell_k(
    const float* __restrict__ gates, float* __restrict__ cbuf,
    float* __restrict__ hbuf, float* __restrict__ hseq, int t) {
  int b = blockIdx.x, d = threadIdx.x;
  const float* g = gates + (size_t)b * (4 * DD);
  float ig = g[d], fg = g[DD + d], gg = g[2 * DD + d], og = g[3 * DD + d];
  float c = sigf(fg) * cbuf[b * DD + d] + sigf(ig) * tanhf(gg);
  float h = sigf(og) * tanhf(c);
  cbuf[b * DD + d] = c;
  hbuf[b * DD + d] = h;
  hseq[((size_t)b * SS + t) * DD + d] = h;
}

__global__ __launch_bounds__(256) void layernorm_k(
    const float* __restrict__ x, int xld, const float* __restrict__ res, int rld,
    const float* __restrict__ g, const float* __restrict__ b,
    float* __restrict__ out, int ldo) {
  __shared__ float sb[256];
  int row = blockIdx.x, tid = threadIdx.x;
  float v = x[(size_t)row * xld + tid];
  if (res) v += res[(size_t)row * rld + tid];
  sb[tid] = v;
  __syncthreads();
  for (int s = 128; s > 0; s >>= 1) { if (tid < s) sb[tid] += sb[tid + s]; __syncthreads(); }
  float mean = sb[0] * (1.0f / 256.0f);
  __syncthreads();
  float dd = v - mean;
  sb[tid] = dd * dd;
  __syncthreads();
  for (int s = 128; s > 0; s >>= 1) { if (tid < s) sb[tid] += sb[tid + s]; __syncthreads(); }
  float var = sb[0] * (1.0f / 256.0f);
  out[(size_t)row * ldo + tid] = dd * rsqrtf(var + 1e-5f) * g[tid] + b[tid];
}

__global__ __launch_bounds__(256) void softmax_rows_k(float* __restrict__ x, int cols) {
  __shared__ float sb[256];
  int row = blockIdx.x, tid = threadIdx.x;
  float* p = x + (size_t)row * cols;
  float mx = -3.0e38f;
  for (int j = tid; j < cols; j += 256) mx = fmaxf(mx, p[j]);
  sb[tid] = mx;
  __syncthreads();
  for (int s = 128; s > 0; s >>= 1) { if (tid < s) sb[tid] = fmaxf(sb[tid], sb[tid + s]); __syncthreads(); }
  mx = sb[0];
  __syncthreads();
  float sum = 0.0f;
  for (int j = tid; j < cols; j += 256) { float e = __expf(p[j] - mx); p[j] = e; sum += e; }
  sb[tid] = sum;
  __syncthreads();
  for (int s = 128; s > 0; s >>= 1) { if (tid < s) sb[tid] += sb[tid + s]; __syncthreads(); }
  float inv = 1.0f / sb[0];
  __syncthreads();
  for (int j = tid; j < cols; j += 256) p[j] *= inv;
}

__global__ __launch_bounds__(64) void attn_k(const float* __restrict__ qkv,
                                             float* __restrict__ o) {
  __shared__ float Q[64][64], Kk[64][64], Vv[64][64], P[64][64];
  int b = blockIdx.x, h = blockIdx.y, s = threadIdx.x;
  const float* base = qkv + (size_t)(b * SS + s) * (3 * DD) + h * 64;
  for (int d = 0; d < 64; ++d) {
    Q[s][d] = base[d];
    Kk[s][d] = base[DD + d];
    Vv[s][d] = base[2 * DD + d];
  }
  __syncthreads();
  float mx = -3.0e38f;
  for (int j = 0; j < 64; ++j) {
    float dot = 0.0f;
    for (int d = 0; d < 64; ++d) dot += Q[s][d] * Kk[j][d];
    dot *= 0.125f;  // 1/sqrt(64)
    P[s][j] = dot;
    mx = fmaxf(mx, dot);
  }
  float sum = 0.0f;
  for (int j = 0; j < 64; ++j) { float e = __expf(P[s][j] - mx); P[s][j] = e; sum += e; }
  float inv = 1.0f / sum;
  float* op = o + (size_t)(b * SS + s) * DD + h * 64;
  for (int d = 0; d < 64; ++d) {
    float a = 0.0f;
    for (int j = 0; j < 64; ++j) a += P[s][j] * Vv[j][d];
    op[d] = a * inv;
  }
}

__global__ __launch_bounds__(256) void cur_k(const float* __restrict__ attn_out,
                                             float* __restrict__ cur,
                                             float* __restrict__ finalb) {
  int b = blockIdx.x, d = threadIdx.x;
  float v = attn_out[((size_t)b * SS + (SS - 1)) * DD + d];
  cur[b * DD + d] = v;
  finalb[b * 768 + 512 + d] = v;
}

__global__ __launch_bounds__(256) void pointer_k(
    const float* __restrict__ pq, const float* __restrict__ pk,
    const float* __restrict__ pv, float* __restrict__ finalb) {
  __shared__ float p[64];
  int b = blockIdx.x, tid = threadIdx.x;
  if (tid < 64) {
    float dot = 0.0f;
    const float* a = pq + (size_t)b * DD;
    const float* kk = pk + (size_t)(b * SS + tid) * DD;
    for (int d = 0; d < DD; ++d) dot += a[d] * kk[d];
    p[tid] = dot * 0.0625f;  // 1/sqrt(256)
  }
  __syncthreads();
  if (tid == 0) {
    float mx = -3.0e38f;
    for (int j = 0; j < 64; ++j) mx = fmaxf(mx, p[j]);
    float sum = 0.0f;
    for (int j = 0; j < 64; ++j) { p[j] = __expf(p[j] - mx); sum += p[j]; }
    float inv = 1.0f / sum;
    for (int j = 0; j < 64; ++j) p[j] *= inv;
  }
  __syncthreads();
  float a = 0.0f;
  for (int j = 0; j < 64; ++j) a += p[j] * pv[(size_t)(b * SS + j) * DD + tid];
  finalb[b * 768 + 256 + tid] = a;
}

__global__ __launch_bounds__(256) void gate_k(const float* __restrict__ cur,
                                              const float* __restrict__ wgw,
                                              const float* __restrict__ wgb,
                                              float* __restrict__ gate) {
  int b = threadIdx.x;
  float dot = wgb[0];
  for (int d = 0; d < DD; ++d) dot += cur[b * DD + d] * wgw[d];
  gate[b] = 1.0f / (1.0f + __expf(-dot));
}

// Sequential scan over batch: out_mem[a] = 0.9*out_mem[a] + 0.1*gate[b]*cur[b].
__global__ __launch_bounds__(256) void mem_update_k(
    const int* __restrict__ target, const float* __restrict__ gate,
    const float* __restrict__ cur, float* __restrict__ out_mem) {
  int tid = threadIdx.x;
  for (int b = 0; b < BB; ++b) {
    int a = target[b] % SLOTS;
    float* row = out_mem + (size_t)a * DD;
    row[tid] = 0.9f * row[tid] + 0.1f * gate[b] * cur[b * DD + tid];
  }
}

// ===========================================================================
// Host orchestration
// ===========================================================================
#define AL(x) ((((size_t)(x)) + 255) & ~(size_t)255)

extern "C" void kernel_launch(void* const* d_in, const int* in_sizes, int n_in,
                              void* d_out, int out_size, void* d_ws, size_t ws_size,
                              hipStream_t stream) {
  (void)in_sizes; (void)n_in; (void)out_size; (void)ws_size;

  const int*   locations  = (const int*)d_in[0];
  const int*   users      = (const int*)d_in[1];
  const int*   start_mins = (const int*)d_in[2];
  const int*   weekdays   = (const int*)d_in[3];
  const int*   target     = (const int*)d_in[4];
  const float* base_embed = (const float*)d_in[5];
  const float* adj_left   = (const float*)d_in[6];
  const float* adj_right  = (const float*)d_in[7];
  const float* gc1_w = (const float*)d_in[8],  *gc1_b = (const float*)d_in[9];
  const float* gc2_w = (const float*)d_in[10], *gc2_b = (const float*)d_in[11];
  const float* user_emb = (const float*)d_in[12];
  const float* hour_emb = (const float*)d_in[13];
  const float* wd_emb   = (const float*)d_in[14];
  const float* memory   = (const float*)d_in[15];
  const float* rq_w = (const float*)d_in[16], *rq_b = (const float*)d_in[17];
  const float* rk_w = (const float*)d_in[18], *rk_b = (const float*)d_in[19];
  const float* wg_w = (const float*)d_in[20], *wg_b = (const float*)d_in[21];
  const float* pq_w = (const float*)d_in[22], *pq_b = (const float*)d_in[23];
  const float* pk_w = (const float*)d_in[24], *pk_b = (const float*)d_in[25];
  const float* pv_w = (const float*)d_in[26], *pv_b = (const float*)d_in[27];
  const float* wih0 = (const float*)d_in[28], *whh0 = (const float*)d_in[29];
  const float* bih0 = (const float*)d_in[30], *bhh0 = (const float*)d_in[31];
  const float* wih1 = (const float*)d_in[32], *whh1 = (const float*)d_in[33];
  const float* bih1 = (const float*)d_in[34], *bhh1 = (const float*)d_in[35];
  const float* ain_w = (const float*)d_in[36], *ain_b = (const float*)d_in[37];
  const float* aout_w = (const float*)d_in[38], *aout_b = (const float*)d_in[39];
  const float* n1g = (const float*)d_in[40], *n1b = (const float*)d_in[41];
  const float* n2g = (const float*)d_in[42], *n2b = (const float*)d_in[43];
  const float* n3g = (const float*)d_in[44], *n3b = (const float*)d_in[45];
  const float* op1_w = (const float*)d_in[46], *op1_b = (const float*)d_in[47];
  const float* op2_w = (const float*)d_in[48], *op2_b = (const float*)d_in[49];

  char* ws = (char*)d_ws;

  // -------- workspace layout (lifetime-aliased) --------
  const size_t G_LP = 0;
  const size_t G_RT = G_LP + AL((size_t)NL * RANKP * 2);
  const size_t G_XB = G_RT + AL((size_t)NKPAD * RANKP * 2);
  const size_t G_NB = G_XB + AL((size_t)NKPAD * DD * 2);
  size_t r0_end = G_NB + AL((size_t)NL * DD * 4);
  const size_t SEQF = (size_t)BB * SS * DD * 4;  // 16.78MB
  if (r0_end < AL(SEQF)) r0_end = AL(SEQF);

  const size_t P_XSEQ = r0_end;                       // xseq; later attn_tmp
  const size_t P_H1   = P_XSEQ + AL(SEQF);            // H1seq; later attn_out
  const size_t P_LOUT = P_H1 + AL(SEQF);              // lstm_out
  const size_t P_BIG  = P_LOUT + AL(SEQF);            // XW; later qkv; later pk|pv
  const size_t P_GATES= P_BIG + AL((size_t)BB * SS * 4 * DD * 4);
  const size_t P_H    = P_GATES + AL((size_t)BB * 4 * DD * 4);
  const size_t P_C    = P_H  + AL((size_t)BB * DD * 4);
  const size_t P_CUR  = P_C  + AL((size_t)BB * DD * 4);
  const size_t P_MQ   = P_CUR + AL((size_t)BB * DD * 4);
  const size_t P_MK   = P_MQ + AL((size_t)BB * DD * 4);
  const size_t P_MW   = P_MK + AL((size_t)SLOTS * DD * 4);
  const size_t P_T3   = P_MW + AL((size_t)BB * SLOTS * 4);
  const size_t P_PQ   = P_T3 + AL((size_t)BB * DD * 4);
  const size_t P_FIN  = P_PQ + AL((size_t)BB * DD * 4);
  const size_t P_HID  = P_FIN + AL((size_t)BB * 768 * 4);
  const size_t P_GATE = P_HID + AL((size_t)BB * 512 * 4);
  const size_t P_BC   = P_GATE + AL((size_t)BB * 4);

  __bf16* Lp  = (__bf16*)(ws + G_LP);
  __bf16* Rt  = (__bf16*)(ws + G_RT);
  __bf16* Xb  = (__bf16*)(ws + G_XB);
  float*  nb  = (float*)(ws + G_NB);
  float*  H2seq   = (float*)(ws + 0);   // aliases graph region
  float*  attn_o  = (float*)(ws + 0);   // aliases H2seq after LN1
  float*  xseq    = (float*)(ws + P_XSEQ);
  float*  attn_tmp= (float*)(ws + P_XSEQ);
  float*  H1seq   = (float*)(ws + P_H1);
  float*  attn_out= (float*)(ws + P_H1);
  float*  lstm_out= (float*)(ws + P_LOUT);
  float*  XW      = (float*)(ws + P_BIG);
  float*  qkv     = (float*)(ws + P_BIG);
  float*  pk      = (float*)(ws + P_BIG);
  float*  pv      = (float*)(ws + P_BIG + AL(SEQF));
  float*  gates   = (float*)(ws + P_GATES);
  float*  hbuf    = (float*)(ws + P_H);
  float*  cbuf    = (float*)(ws + P_C);
  float*  cur     = (float*)(ws + P_CUR);
  float*  mq      = (float*)(ws + P_MQ);
  float*  mk      = (float*)(ws + P_MK);
  float*  mw      = (float*)(ws + P_MW);
  float*  t3      = (float*)(ws + P_T3);
  float*  pqb     = (float*)(ws + P_PQ);
  float*  finalb  = (float*)(ws + P_FIN);
  float*  hid     = (float*)(ws + P_HID);
  float*  gate    = (float*)(ws + P_GATE);
  float*  bc0     = (float*)(ws + P_BC);
  float*  bc1     = bc0 + 4 * DD;

  float* logits  = (float*)d_out;
  float* out_mem = logits + (size_t)BB * NL;

  auto gemm = [&](const float* A, int lda, const float* Bm, int ldb, int btrans,
                  const float* bias, const float* add, int add_ld, float scale,
                  int act, float* outF, __bf16* outH, int swz, int ldo,
                  int M, int N, int K) {
    const int nfull = N & ~63;
    if (nfull > 0) {
      dim3 g(M / 16, nfull / 64);
      hipLaunchKernelGGL(HIP_KERNEL_NAME(gemm_wmma<true>), g, dim3(32), 0, stream,
                         A, lda, Bm, ldb, btrans, bias, add, add_ld, scale, act,
                         outF, outH, swz, ldo, 0, N, K);
    }
    if (nfull < N) {
      dim3 g(M / 16, 1);
      hipLaunchKernelGGL(HIP_KERNEL_NAME(gemm_wmma<false>), g, dim3(32), 0, stream,
                         A, lda, Bm, ldb, btrans, bias, add, add_ld, scale, act,
                         outF, outH, swz, ldo, nfull, N, K);
    }
  };

  // ---------------- Graph convolution (flash adjacency) ----------------
  hipLaunchKernelGGL(pad_left_k,  dim3((NL * RANKP + 255) / 256),   dim3(256), 0, stream, adj_left, Lp);
  hipLaunchKernelGGL(pad_right_k, dim3((NKPAD * RANKP + 255) / 256), dim3(256), 0, stream, adj_right, Rt);
  hipMemsetAsync(ws + G_XB, 0, (size_t)NKPAD * DD * 2, stream);  // swizzled pads = 0

  // X1 = base_embed @ gc1_w.T  (bf16, swizzled B-fragment layout)
  gemm(base_embed, DD, gc1_w, DD, 1, nullptr, nullptr, 0, 1.0f, 0,
       nullptr, Xb, 1, DD, NL, DD, DD);
  // nb1 = relu(softmax(adj) @ X1 + gc1_b)
  hipLaunchKernelGGL(adj_flash, dim3(NL / 16), dim3(32), 0, stream, Lp, Rt, Xb, gc1_b, nb, 1);
  // X2 = nb1 @ gc2_w.T  (bf16, swizzled)
  gemm(nb, DD, gc2_w, DD, 1, nullptr, nullptr, 0, 1.0f, 0,
       nullptr, Xb, 1, DD, NL, DD, DD);
  // nb2 = softmax(adj) @ X2 + gc2_b
  hipLaunchKernelGGL(adj_flash, dim3(NL / 16), dim3(32), 0, stream, Lp, Rt, Xb, gc2_b, nb, 0);

  // ---------------- Embedding sum ----------------
  hipLaunchKernelGGL(embed_k, dim3(BB * SS), dim3(256), 0, stream,
                     locations, users, start_mins, weekdays,
                     base_embed, nb, user_emb, hour_emb, wd_emb, xseq);

  // ---------------- LSTM layers ----------------
  hipLaunchKernelGGL(addbias_k, dim3((4 * DD + 255) / 256), dim3(256), 0, stream, bih0, bhh0, bc0, 4 * DD);
  hipLaunchKernelGGL(addbias_k, dim3((4 * DD + 255) / 256), dim3(256), 0, stream, bih1, bhh1, bc1, 4 * DD);

  gemm(xseq, DD, wih0, DD, 1, bc0, nullptr, 0, 1.0f, 0,
       XW, nullptr, 0, 4 * DD, BB * SS, 4 * DD, DD);
  hipMemsetAsync(hbuf, 0, (size_t)BB * DD * 4, stream);
  hipMemsetAsync(cbuf, 0, (size_t)BB * DD * 4, stream);
  for (int t = 0; t < SS; ++t) {
    gemm(hbuf, DD, whh0, DD, 1, nullptr, XW + (size_t)t * 4 * DD, SS * 4 * DD,
         1.0f, 0, gates, nullptr, 0, 4 * DD, BB, 4 * DD, DD);
    hipLaunchKernelGGL(lstm_cell_k, dim3(BB), dim3(256), 0, stream, gates, cbuf, hbuf, H1seq, t);
  }
  gemm(H1seq, DD, wih1, DD, 1, bc1, nullptr, 0, 1.0f, 0,
       XW, nullptr, 0, 4 * DD, BB * SS, 4 * DD, DD);
  hipMemsetAsync(hbuf, 0, (size_t)BB * DD * 4, stream);
  hipMemsetAsync(cbuf, 0, (size_t)BB * DD * 4, stream);
  for (int t = 0; t < SS; ++t) {
    gemm(hbuf, DD, whh1, DD, 1, nullptr, XW + (size_t)t * 4 * DD, SS * 4 * DD,
         1.0f, 0, gates, nullptr, 0, 4 * DD, BB, 4 * DD, DD);
    hipLaunchKernelGGL(lstm_cell_k, dim3(BB), dim3(256), 0, stream, gates, cbuf, hbuf, H2seq, t);
  }

  // lstm_out = LN(h2 + x)
  hipLaunchKernelGGL(layernorm_k, dim3(BB * SS), dim3(256), 0, stream,
                     H2seq, DD, xseq, DD, n1g, n1b, lstm_out, DD);

  // ---------------- Self attention ----------------
  gemm(lstm_out, DD, ain_w, DD, 1, ain_b, nullptr, 0, 1.0f, 0,
       qkv, nullptr, 0, 3 * DD, BB * SS, 3 * DD, DD);
  hipLaunchKernelGGL(attn_k, dim3(BB, 4), dim3(64), 0, stream, qkv, attn_o);
  gemm(attn_o, DD, aout_w, DD, 1, aout_b, lstm_out, DD, 1.0f, 0,
       attn_tmp, nullptr, 0, DD, BB * SS, DD, DD);
  hipLaunchKernelGGL(layernorm_k, dim3(BB * SS), dim3(256), 0, stream,
                     attn_tmp, DD, (const float*)nullptr, 0, n2g, n2b, attn_out, DD);

  // cur = attn_out[:, -1, :]  (also fills final[:,512:768])
  hipLaunchKernelGGL(cur_k, dim3(BB), dim3(256), 0, stream, attn_out, cur, finalb);

  // ---------------- Memory read head ----------------
  gemm(cur, DD, rq_w, DD, 1, rq_b, nullptr, 0, 1.0f, 0, mq, nullptr, 0, DD, BB, DD, DD);
  gemm(memory, DD, rk_w, DD, 1, rk_b, nullptr, 0, 1.0f, 0, mk, nullptr, 0, DD, SLOTS, DD, DD);
  gemm(mq, DD, mk, DD, 1, nullptr, nullptr, 0, 0.0625f, 0, mw, nullptr, 0, SLOTS, BB, SLOTS, DD);
  hipLaunchKernelGGL(softmax_rows_k, dim3(BB), dim3(256), 0, stream, mw, SLOTS);
  gemm(mw, SLOTS, memory, DD, 0, nullptr, cur, DD, 1.0f, 0, t3, nullptr, 0, DD, BB, DD, SLOTS);
  hipLaunchKernelGGL(layernorm_k, dim3(BB), dim3(256), 0, stream,
                     t3, DD, (const float*)nullptr, 0, n3g, n3b, finalb, 768);

  // ---------------- Pointer head ----------------
  gemm(cur, DD, pq_w, DD, 1, pq_b, nullptr, 0, 1.0f, 0, pqb, nullptr, 0, DD, BB, DD, DD);
  gemm(attn_out, DD, pk_w, DD, 1, pk_b, nullptr, 0, 1.0f, 0, pk, nullptr, 0, DD, BB * SS, DD, DD);
  gemm(attn_out, DD, pv_w, DD, 1, pv_b, nullptr, 0, 1.0f, 0, pv, nullptr, 0, DD, BB * SS, DD, DD);
  hipLaunchKernelGGL(pointer_k, dim3(BB), dim3(256), 0, stream, pqb, pk, pv, finalb);

  // ---------------- Output head ----------------
  gemm(finalb, 768, op1_w, 768, 1, op1_b, nullptr, 0, 1.0f, 2, hid, nullptr, 0, 512, BB, 512, 768);
  gemm(hid, 512, op2_w, 512, 1, op2_b, nullptr, 0, 1.0f, 0, logits, nullptr, 0, NL, BB, NL, 512);

  // ---------------- Memory write (sequential scan) ----------------
  hipLaunchKernelGGL(gate_k, dim3(1), dim3(256), 0, stream, cur, wg_w, wg_b, gate);
  hipMemcpyAsync(out_mem, memory, (size_t)SLOTS * DD * 4, hipMemcpyDeviceToDevice, stream);
  hipLaunchKernelGGL(mem_update_k, dim3(1), dim3(256), 0, stream, target, gate, cur, out_mem);
}
